// Qwen3TTSAttention_71141838291451
// MI455X (gfx1250) — matrix operand order repaired and checked by reading, not compiled, not run on hardware
//
#include <hip/hip_runtime.h>
#include <hip/hip_bf16.h>

// ---------------------------------------------------------------------------
// Qwen3-TTS attention layer for MI455X (gfx1250, wave32, WMMA + TDM).
// Pipeline:
//   gemm_wmma_f16 (q|k|v, f32)  ->  rmsnorm_rope_f16 (q,k -> f16 bufs)
//   v_transpose_f16 (v -> f16, (kv,d,t))  ->  flash_attn (TDM tiles + WMMA)
//   gemm_wmma_f16 (o-proj)
// Matrix math: v_wmma_f32_16x16x32_f16. Attention tiles: tensor_load_to_lds.
// ---------------------------------------------------------------------------

typedef _Float16     h8    __attribute__((ext_vector_type(8)));
typedef _Float16     v16h  __attribute__((ext_vector_type(16)));
typedef float        v8f   __attribute__((ext_vector_type(8)));
typedef unsigned int u32;
typedef u32          u32x4 __attribute__((ext_vector_type(4)));
typedef int          i32x4 __attribute__((ext_vector_type(4)));
typedef int          i32x8 __attribute__((ext_vector_type(8)));

#define TSEQ     4096
#define HIDDEN   2048
#define NHEADS   16
#define NKV      4
#define GRP      4
#define HDIM     128
#define ATT_SCALE 0.08838834764831845f  // 128^-0.5
#define LOG_THETA 9.210340371976184f    // ln(10000)

#ifndef __has_builtin
#define __has_builtin(x) 0
#endif
#if __has_builtin(__builtin_amdgcn_tensor_load_to_lds) && \
    __has_builtin(__builtin_amdgcn_s_wait_tensorcnt)
#define USE_TDM 1
#else
#define USE_TDM 0
#endif

union Frag { v16h v; h8 h[2]; };

#if USE_TDM
// 2D Tensor-DMA load: tile (tile_d0 x tile_d1) of f16 from global into LDS,
// walking dim0 (contiguous in memory) fastest; LDS written sequentially.
// Descriptor packing per CDNA5 ISA ch.8 (D# groups 0/1; groups 2/3 unused).
// This toolchain exposes the 6-arg builtin:
//   (u32x4 g0, i32x8 g1, i32x4 g2, i32x4 g3, i32x8 extra, i32 cpol)
__device__ __forceinline__ void tdm_load_2d_f16(u32 lds_off, const void* gptr,
                                                u32 tensor_d0, u32 tensor_d1,
                                                u32 tile_d0, u32 tile_d1,
                                                u32 stride0_elems) {
  unsigned long long ga = (unsigned long long)gptr;
  u32x4 g0;
  g0.x = 1u;                                   // count=1, user descriptor
  g0.y = lds_off;                              // lds_addr (bytes)
  g0.z = (u32)(ga & 0xffffffffu);              // global_addr[31:0]
  g0.w = (u32)((ga >> 32) & 0x01ffffffu) | (2u << 30);  // addr[56:32] | type=2
  i32x8 g1;
  g1[0] = (int)(1u << 16);                     // data_size=1 (2B); no multicast
  g1[1] = (int)((tensor_d0 & 0xffffu) << 16);  // tensor_dim0[15:0] @ bits63:48
  g1[2] = (int)(((tensor_d0 >> 16) & 0xffffu) | ((tensor_d1 & 0xffffu) << 16));
  g1[3] = (int)(((tensor_d1 >> 16) & 0xffffu) | ((tile_d0 & 0xffffu) << 16));
  g1[4] = (int)(tile_d1 & 0xffffu);            // tile_dim1; tile_dim2 = 0
  g1[5] = (int)stride0_elems;                  // tensor_dim0_stride[31:0]
  g1[6] = 0;                                   // stride0[47:32], stride1[15:0]
  g1[7] = 0;                                   // stride1[47:16]
  i32x4 z4 = {0, 0, 0, 0};
  i32x8 z8 = {0, 0, 0, 0, 0, 0, 0, 0};
  __builtin_amdgcn_tensor_load_to_lds(g0, g1, z4, z4, z8, 0);
}
#endif

// ---------------------------------------------------------------------------
// GEMM: C[M,N] = A[M,K] * B[K,N], f32 in/out, f16 WMMA compute.
// Block: 256 threads (8 waves). Tile: BM=128, BN=64, BK=32.
// Wave w: m-tiles = (w&3)*32 + {0,16}; n-tiles = (w>>2)*32 + {0,16}.
// 4 WMMAs per wave per K-step (2 A-frags x 2 B-frags).
// ---------------------------------------------------------------------------
__global__ __launch_bounds__(256) void gemm_wmma_f16(
    const float* __restrict__ A, const float* __restrict__ B,
    float* __restrict__ C, int M, int N, int K) {
  constexpr int BM = 128, BN = 64, BK = 32;
  constexpr int LAS = 40;  // lds row strides (halves), multiple of 8
  constexpr int LBS = 40;
  __shared__ __align__(16) _Float16 lA[BM * LAS];
  __shared__ __align__(16) _Float16 lB[BN * LBS];
  (void)M;

  const int tid  = threadIdx.x;
  const int lane = tid & 31;
  const int w    = tid >> 5;
  const int m0   = blockIdx.y * BM;
  const int n0   = blockIdx.x * BN;
  const int mi   = (w & 3) * 32;
  const int nb   = (w >> 2) * 32;

  v8f c00 = {}, c01 = {}, c10 = {}, c11 = {};

  // staging maps: A: 128 rows x 2 chunks of 16; B: 32 rows x 8 chunks of 8
  const int am = tid >> 1, ak = (tid & 1) * 16;
  const int bk = tid >> 3, bn = (tid & 7) * 8;

  for (int k0 = 0; k0 < K; k0 += BK) {
    const float* ap = A + (size_t)(m0 + am) * K + k0 + ak;
    const float* bp = B + (size_t)(k0 + bk) * N + n0 + bn;
    __builtin_prefetch(ap + BK, 0, 1);             // global_prefetch_b8
    __builtin_prefetch(bp + (size_t)BK * N, 0, 1);
#pragma unroll
    for (int i = 0; i < 16; ++i) lA[am * LAS + ak + i] = (_Float16)ap[i];
#pragma unroll
    for (int i = 0; i < 8; ++i) lB[(bn + i) * LBS + bk] = (_Float16)bp[i];
    __syncthreads();

    const int arow = mi + (lane & 15);
    const int akb  = (lane >> 4) * 8;
    Frag a0, a1, b0, b1;
    a0.h[0] = *(const h8*)&lA[arow * LAS + akb];
    a0.h[1] = *(const h8*)&lA[arow * LAS + 16 + akb];
    a1.h[0] = *(const h8*)&lA[(arow + 16) * LAS + akb];
    a1.h[1] = *(const h8*)&lA[(arow + 16) * LAS + 16 + akb];
    const int bnr = nb + (lane & 15);
    const int bkb = (lane >> 4) * 8;
    b0.h[0] = *(const h8*)&lB[bnr * LBS + bkb];
    b0.h[1] = *(const h8*)&lB[bnr * LBS + bkb + 16];
    b1.h[0] = *(const h8*)&lB[(bnr + 16) * LBS + bkb];
    b1.h[1] = *(const h8*)&lB[(bnr + 16) * LBS + bkb + 16];

    c00 = __builtin_amdgcn_wmma_f32_16x16x32_f16(false, a0.v, false, b0.v,
                                                 (short)0, c00, false, false);
    c01 = __builtin_amdgcn_wmma_f32_16x16x32_f16(false, a0.v, false, b1.v,
                                                 (short)0, c01, false, false);
    c10 = __builtin_amdgcn_wmma_f32_16x16x32_f16(false, a1.v, false, b0.v,
                                                 (short)0, c10, false, false);
    c11 = __builtin_amdgcn_wmma_f32_16x16x32_f16(false, a1.v, false, b1.v,
                                                 (short)0, c11, false, false);
    __syncthreads();
  }

  // C layout: VGPR r holds row m = r + 8*(lane>>4); col = lane&15
  const int mrow = m0 + mi + (lane >> 4) * 8;
  const int col0 = n0 + nb + (lane & 15);
#pragma unroll
  for (int r = 0; r < 8; ++r) {
    C[(size_t)(mrow + r) * N + col0]           = c00[r];
    C[(size_t)(mrow + r) * N + col0 + 16]      = c01[r];
    C[(size_t)(mrow + r + 16) * N + col0]      = c10[r];
    C[(size_t)(mrow + r + 16) * N + col0 + 16] = c11[r];
  }
}

// ---------------------------------------------------------------------------
// Fused per-head RMSNorm + RoPE; reads f32, writes f16 (same layout).
// 64 threads per (t, head) row; thread d owns rotation pair (d, d+64).
// ---------------------------------------------------------------------------
__global__ __launch_bounds__(64) void rmsnorm_rope_f16(
    const float* __restrict__ xin, _Float16* __restrict__ xout,
    const float* __restrict__ wn, const int* __restrict__ positions,
    int n_heads) {
  const int row = blockIdx.x;           // t*n_heads + head
  const int t   = row / n_heads;
  const int d   = threadIdx.x;          // 0..63
  const float* p = xin + (size_t)row * HDIM;
  _Float16* po = xout + (size_t)row * HDIM;

  float x1 = p[d], x2 = p[d + 64];
  __shared__ float red[64];
  red[d] = x1 * x1 + x2 * x2;
  __syncthreads();
  for (int s = 32; s > 0; s >>= 1) {
    if (d < s) red[d] += red[d + s];
    __syncthreads();
  }
  const float rinv = rsqrtf(red[0] * (1.0f / HDIM) + 1e-6f);
  const float n1 = x1 * rinv * wn[d];
  const float n2 = x2 * rinv * wn[d + 64];

  const float pos = (float)positions[t];
  const float inv_freq = __expf(-(float)d * (1.0f / 64.0f) * LOG_THETA);
  const float fr = pos * inv_freq;
  const float c = __cosf(fr), s = __sinf(fr);
  po[d]      = (_Float16)(n1 * c - n2 * s);
  po[d + 64] = (_Float16)(n2 * c + n1 * s);
}

// ---------------------------------------------------------------------------
// V: (T, NKV*128) f32 -> vt: (NKV*128, T) f16, so TDM dim0 (contiguous) walks
// the key axis and lands [d][key] tiles in LDS directly.
// ---------------------------------------------------------------------------
__global__ __launch_bounds__(256) void v_transpose_f16(
    const float* __restrict__ v, _Float16* __restrict__ vt) {
  const int idx = blockIdx.x * 256 + threadIdx.x;
  const int t = idx >> 9;        // T index
  const int c = idx & 511;       // kv*128 + d
  vt[(size_t)c * TSEQ + t] = (_Float16)v[idx];
}

// ---------------------------------------------------------------------------
// Flash attention (causal, GQA). Block = 256 threads = 8 waves.
// Block: 128 query rows for one (kv head, group); each wave owns 16 rows.
// Tiles arrive via Tensor Data Mover (wave 0 issues, TENSORcnt wait, barrier)
// or fallback VGPR staging. Scores + PV via WMMA f16->f32.
// ---------------------------------------------------------------------------
__global__ __launch_bounds__(256) void flash_attn(
    const _Float16* __restrict__ qh, const _Float16* __restrict__ kh,
    const _Float16* __restrict__ vt, float* __restrict__ ob) {
  constexpr int QS = 128;  // lQ row stride (halves): TDM writes packed rows
  constexpr int KS = 128;  // lK row stride
  constexpr int VS = 32;   // lVt row stride (32 keys)
  constexpr int PS = 40;   // lP row stride
  __shared__ __align__(16) _Float16 lQ[128 * QS];    // [qrow][d]
  __shared__ __align__(16) _Float16 lK[32 * KS];     // [key][d]
  __shared__ __align__(16) _Float16 lVt[128 * VS];   // [d][key]
  __shared__ __align__(16) _Float16 lP[8 * 16 * PS]; // per-wave P [m][key]

  const int tid  = threadIdx.x;
  const int lane = tid & 31;
  const int w    = tid >> 5;
  const int kk   = blockIdx.y >> 2;
  const int g    = blockIdx.y & 3;
  const int h    = kk * GRP + g;
  const int qbase = blockIdx.x * 128;

#if USE_TDM
  if (w == 0) {
    // Q tile: 128 rows x 128 d, row stride NHEADS*HDIM halves
    tdm_load_2d_f16((u32)(unsigned long long)(void*)lQ,
                    qh + (size_t)qbase * (NHEADS * HDIM) + h * HDIM,
                    HDIM, TSEQ, HDIM, 128, NHEADS * HDIM);
  }
#else
#pragma unroll 4
  for (int i = 0; i < 64; ++i) {
    int e = i * 256 + tid, row = e >> 7, d = e & 127;
    lQ[row * QS + d] = qh[(size_t)(qbase + row) * (NHEADS * HDIM) + h * HDIM + d];
  }
#endif

  const int m0   = w * 16;            // wave's query rows inside block
  const int rgrp = (lane >> 4) * 8;   // C-layout row group offset
  const int ncol = lane & 15;

  float mr[8], lr[8];
  v8f o[8];
#pragma unroll
  for (int r = 0; r < 8; ++r) { mr[r] = -3.0e38f; lr[r] = 0.0f; }
#pragma unroll
  for (int t = 0; t < 8; ++t) { v8f z = {}; o[t] = z; }

  for (int j0 = 0; j0 < qbase + 128; j0 += 32) {
#if USE_TDM
    if (w == 0) {
      // K tile: 32 keys x 128 d -> lK[key][d]
      tdm_load_2d_f16((u32)(unsigned long long)(void*)lK,
                      kh + (size_t)j0 * (NKV * HDIM) + kk * HDIM,
                      HDIM, TSEQ, HDIM, 32, NKV * HDIM);
      // V tile: 128 d x 32 keys -> lVt[d][key] (vt is (kv*128, T))
      tdm_load_2d_f16((u32)(unsigned long long)(void*)lVt,
                      vt + (size_t)kk * HDIM * TSEQ + j0,
                      TSEQ, HDIM, 32, HDIM, TSEQ);
      __builtin_amdgcn_s_wait_tensorcnt((short)0);
    }
#else
#pragma unroll 4
    for (int i = 0; i < 16; ++i) {
      int e = i * 256 + tid, key = e >> 7, d = e & 127;
      lK[key * KS + d]  = kh[(size_t)(j0 + key) * (NKV * HDIM) + kk * HDIM + d];
      lVt[d * VS + key] = vt[(size_t)(kk * HDIM + d) * TSEQ + j0 + key];
    }
#endif
    __syncthreads();

    // ---- scores S(16x32) = Q(16x128) . K^T ----
    v8f s0 = {};
    v8f s1 = {};
    const int arow = m0 + (lane & 15);
    const int akb  = (lane >> 4) * 8;
    const int bkey = lane & 15;
    const int bkb  = (lane >> 4) * 8;
#pragma unroll
    for (int ds = 0; ds < 4; ++ds) {
      Frag af, b0, b1;
      af.h[0] = *(const h8*)&lQ[arow * QS + ds * 32 + akb];
      af.h[1] = *(const h8*)&lQ[arow * QS + ds * 32 + 16 + akb];
      b0.h[0] = *(const h8*)&lK[bkey * KS + ds * 32 + bkb];
      b0.h[1] = *(const h8*)&lK[bkey * KS + ds * 32 + bkb + 16];
      b1.h[0] = *(const h8*)&lK[(bkey + 16) * KS + ds * 32 + bkb];
      b1.h[1] = *(const h8*)&lK[(bkey + 16) * KS + ds * 32 + bkb + 16];
      s0 = __builtin_amdgcn_wmma_f32_16x16x32_f16(false, af.v, false, b0.v,
                                                  (short)0, s0, false, false);
      s1 = __builtin_amdgcn_wmma_f32_16x16x32_f16(false, af.v, false, b1.v,
                                                  (short)0, s1, false, false);
    }

    // ---- online softmax (row spans one 16-lane half; VGPR r = row) ----
#pragma unroll
    for (int r = 0; r < 8; ++r) {
      const int m    = m0 + rgrp + r;
      const int qidx = qbase + m;
      const int key0 = j0 + ncol;
      const int key1 = key0 + 16;
      const bool ok0 = (key0 <= qidx), ok1 = (key1 <= qidx);
      float v0 = ok0 ? s0[r] * ATT_SCALE : -3.0e38f;
      float v1 = ok1 ? s1[r] * ATT_SCALE : -3.0e38f;
      float rmax = fmaxf(v0, v1);
      for (int off = 1; off < 16; off <<= 1)
        rmax = fmaxf(rmax, __shfl_xor(rmax, off, 32));
      const float mnew  = fmaxf(mr[r], rmax);
      const float alpha = __expf(mr[r] - mnew);
      const float e0 = ok0 ? __expf(v0 - mnew) : 0.0f;
      const float e1 = ok1 ? __expf(v1 - mnew) : 0.0f;
      float rsum = e0 + e1;
      for (int off = 1; off < 16; off <<= 1)
        rsum += __shfl_xor(rsum, off, 32);
      lr[r] = lr[r] * alpha + rsum;
      mr[r] = mnew;
#pragma unroll
      for (int t = 0; t < 8; ++t) o[t][r] *= alpha;
      const int mw = rgrp + r;  // wave-local row, C-layout -> A-layout via LDS
      lP[(w * 16 + mw) * PS + ncol]      = (_Float16)e0;
      lP[(w * 16 + mw) * PS + ncol + 16] = (_Float16)e1;
    }

    // ---- O(16x128) += P(16x32) . V(32x128) ----
    Frag pf;
    const int pm  = lane & 15;
    const int pkb = (lane >> 4) * 8;
    pf.h[0] = *(const h8*)&lP[(w * 16 + pm) * PS + pkb];
    pf.h[1] = *(const h8*)&lP[(w * 16 + pm) * PS + 16 + pkb];
    const int vkb = (lane >> 4) * 8;
#pragma unroll
    for (int t = 0; t < 8; ++t) {
      Frag bf;
      const int n = t * 16 + (lane & 15);  // d column
      bf.h[0] = *(const h8*)&lVt[n * VS + vkb];
      bf.h[1] = *(const h8*)&lVt[n * VS + vkb + 16];
      o[t] = __builtin_amdgcn_wmma_f32_16x16x32_f16(false, pf.v, false, bf.v,
                                                    (short)0, o[t], false, false);
    }
    __syncthreads();
  }

  // normalize and write out: col = h*128 + d
#pragma unroll
  for (int r = 0; r < 8; ++r) {
    const int m = m0 + rgrp + r;
    const float inv = 1.0f / lr[r];
#pragma unroll
    for (int t = 0; t < 8; ++t) {
      ob[(size_t)(qbase + m) * (NHEADS * HDIM) + h * HDIM + t * 16 + ncol] =
          o[t][r] * inv;
    }
  }
}

// ---------------------------------------------------------------------------
// Host-side launch. d_in: positions, hidden_states, wq, wk, wv, wo,
// q_norm_w, k_norm_w. d_out: (T, HIDDEN) f32.
// Workspace (floats then halves):
//   qf (T*2048) | kf (T*512) | vf (T*512) | attn (T*2048)   f32
//   qh (T*2048) | kh (T*512) | vth (T*512)                  f16
// ---------------------------------------------------------------------------
extern "C" void kernel_launch(void* const* d_in, const int* in_sizes, int n_in,
                              void* d_out, int out_size, void* d_ws,
                              size_t ws_size, hipStream_t stream) {
  (void)in_sizes; (void)n_in; (void)out_size; (void)ws_size;
  const int*   positions = (const int*)d_in[0];
  const float* hs = (const float*)d_in[1];
  const float* wq = (const float*)d_in[2];
  const float* wk = (const float*)d_in[3];
  const float* wv = (const float*)d_in[4];
  const float* wo = (const float*)d_in[5];
  const float* qnw = (const float*)d_in[6];
  const float* knw = (const float*)d_in[7];
  float* out = (float*)d_out;

  float* qf   = (float*)d_ws;
  float* kf   = qf + (size_t)TSEQ * NHEADS * HDIM;
  float* vf   = kf + (size_t)TSEQ * NKV * HDIM;
  float* attn = vf + (size_t)TSEQ * NKV * HDIM;
  _Float16* qh  = (_Float16*)(attn + (size_t)TSEQ * NHEADS * HDIM);
  _Float16* kh  = qh + (size_t)TSEQ * NHEADS * HDIM;
  _Float16* vth = kh + (size_t)TSEQ * NKV * HDIM;

  const dim3 blk(256);

  // QKV projections (f16 WMMA, f32 accumulate)
  gemm_wmma_f16<<<dim3((NHEADS * HDIM) / 64, TSEQ / 128), blk, 0, stream>>>(
      hs, wq, qf, TSEQ, NHEADS * HDIM, HIDDEN);
  gemm_wmma_f16<<<dim3((NKV * HDIM) / 64, TSEQ / 128), blk, 0, stream>>>(
      hs, wk, kf, TSEQ, NKV * HDIM, HIDDEN);
  gemm_wmma_f16<<<dim3((NKV * HDIM) / 64, TSEQ / 128), blk, 0, stream>>>(
      hs, wv, vf, TSEQ, NKV * HDIM, HIDDEN);

  // RMSNorm + RoPE -> f16 tensors for the attention core
  rmsnorm_rope_f16<<<TSEQ * NHEADS, 64, 0, stream>>>(qf, qh, qnw, positions,
                                                     NHEADS);
  rmsnorm_rope_f16<<<TSEQ * NKV, 64, 0, stream>>>(kf, kh, knw, positions, NKV);
  // V -> f16 transposed (kv*128, T) for TDM-friendly [d][key] tiles
  v_transpose_f16<<<(TSEQ * NKV * HDIM) / 256, blk, 0, stream>>>(vf, vth);

  // Causal GQA flash attention: grid (q tiles of 128, 16 head pairs)
  flash_attn<<<dim3(TSEQ / 128, NKV * GRP), blk, 0, stream>>>(qh, kh, vth, attn);

  // Output projection
  gemm_wmma_f16<<<dim3(HIDDEN / 64, TSEQ / 128), blk, 0, stream>>>(
      attn, wo, out, TSEQ, HIDDEN, HIDDEN);
}
